// ModulatedDeformConv2d_6966436954254
// MI455X (gfx1250) — matrix operand order repaired and checked by reading, not compiled, not run on hardware
//
#include <hip/hip_runtime.h>
#include <hip/hip_bf16.h>

typedef __attribute__((ext_vector_type(16))) _Float16 v16h;
typedef __attribute__((ext_vector_type(8)))  _Float16 v8h;
typedef __attribute__((ext_vector_type(4)))  _Float16 v4h;
typedef __attribute__((ext_vector_type(8)))  float    v8f;

#define BB   4
#define CC   64
#define HH   96
#define WW   96
#define OO   64
#define KT   9            // 3x3 taps
#define CK   576          // CC*KT (permuted as k*64+c)
#define HW   9216         // HH*WW
#define NPIX 32           // pixels per workgroup tile
#define KCH  32           // K-chunk per WMMA step
#define NIT  18           // CK/KCH

// ---------------------------------------------------------------------------
// Prep 1: weight f32 [O][c*9+k]  ->  f16 [O][k*64+c]   (72 KB, L2 resident)
// ---------------------------------------------------------------------------
__global__ __launch_bounds__(256) void wprep(const float* __restrict__ w,
                                             _Float16* __restrict__ wh) {
    int i = blockIdx.x * 256 + threadIdx.x;          // over O*CK = 36864
    if (i < OO * CK) {
        int o  = i / CK;
        int ck = i - o * CK;
        int k  = ck >> 6;          // tap
        int c  = ck & 63;          // channel
        wh[i] = (_Float16)w[o * CK + c * KT + k];
    }
}

// ---------------------------------------------------------------------------
// Prep 2: x NCHW f32 -> NHWC f16 (tiled transpose through LDS, fully coalesced)
// ---------------------------------------------------------------------------
__global__ __launch_bounds__(256) void xprep(const float* __restrict__ x,
                                             _Float16* __restrict__ xt) {
    __shared__ float tile[64 * 65];
    int bid = blockIdx.x;                 // BB * (HW/64) = 576 blocks
    int b   = bid / (HW / 64);
    int p0  = (bid % (HW / 64)) * 64;
    int tid = threadIdx.x;
    for (int f = tid; f < 4096; f += 256) {          // read coalesced over p
        int c = f >> 6, pl = f & 63;
        tile[c * 65 + pl] = x[(size_t)(b * CC + c) * HW + p0 + pl];
    }
    __syncthreads();
    for (int f = tid; f < 4096; f += 256) {          // write coalesced over c
        int pl = f >> 6, c = f & 63;
        xt[((size_t)(b * HW + p0 + pl)) * CC + c] = (_Float16)tile[c * 65 + pl];
    }
}

// ---------------------------------------------------------------------------
// Main: deformable gather -> LDS f16 tile (double-buffered) -> WMMA f16->f32
// grid = BB * HH * (WW/NPIX) = 1152 workgroups of 256 threads (8 waves)
// ---------------------------------------------------------------------------
__global__ __launch_bounds__(256) void dcn_wmma(
        const _Float16* __restrict__ xt,      // NHWC f16
        const float*    __restrict__ offset,  // (B,18,H,W)
        const float*    __restrict__ mask,    // (B,9,H,W)
        const _Float16* __restrict__ wh,      // f16 weights [O][k*64+c]
        const float*    __restrict__ bias,
        float*          __restrict__ out) {   // (B,O,H,W)

    __shared__ __align__(64) _Float16 s_lds[2][NPIX * KCH]; // double-buffered tile
    __shared__ int4   meta_i[NPIX * KT];     // 4 gather offsets, pre-scaled by CC
    __shared__ float4 meta_w[NPIX * KT];     // 4 fused bilinear*valid*mask weights

    const int tid = threadIdx.x;
    const int wg  = blockIdx.x;
    const int b   = wg / (HH * (WW / NPIX));
    const int r   = wg % (HH * (WW / NPIX));
    const int ho  = r / (WW / NPIX);
    const int wo0 = (r % (WW / NPIX)) * NPIX;

    // ---- per-(pixel,tap) bilinear metadata (shared across all 64 channels) ----
    for (int t = tid; t < NPIX * KT; t += 256) {
        int n  = t / KT;
        int k  = t - n * KT;
        int wo = wo0 + n;
        int ob = ((b * 18 + 2 * k) * HH + ho) * WW + wo;
        float oy = offset[ob];
        float ox = offset[ob + HW];                        // channel 2k+1
        float mk = mask[((b * KT + k) * HH + ho) * WW + wo];
        float py = (float)(ho - 1 + (k / 3)) + oy;
        float px = (float)(wo - 1 + (k % 3)) + ox;
        float fy = floorf(py), fx = floorf(px);
        float ly = py - fy,    lx = px - fx;
        int y0 = (int)fy, x0 = (int)fx, y1 = y0 + 1, x1 = x0 + 1;
        float vy0 = (y0 >= 0 && y0 < HH) ? 1.f : 0.f;
        float vy1 = (y1 >= 0 && y1 < HH) ? 1.f : 0.f;
        float vx0 = (x0 >= 0 && x0 < WW) ? 1.f : 0.f;
        float vx1 = (x1 >= 0 && x1 < WW) ? 1.f : 0.f;
        int cy0 = min(max(y0, 0), HH - 1), cy1 = min(max(y1, 0), HH - 1);
        int cx0 = min(max(x0, 0), WW - 1), cx1 = min(max(x1, 0), WW - 1);
        meta_i[t] = make_int4((cy0 * WW + cx0) * CC, (cy0 * WW + cx1) * CC,
                              (cy1 * WW + cx0) * CC, (cy1 * WW + cx1) * CC);
        meta_w[t] = make_float4((1.f - ly) * (1.f - lx) * vy0 * vx0 * mk,
                                (1.f - ly) * lx         * vy0 * vx1 * mk,
                                ly         * (1.f - lx) * vy1 * vx0 * mk,
                                ly         * lx         * vy1 * vx1 * mk);
    }

    // ---- static role assignment ----
    const int wv       = tid >> 5;
    const int lane     = tid & 31;
    const int mt       = wv & 3;            // O-tile (4 x 16)
    const int nt       = wv >> 2;           // pixel-tile (2 x 16)
    const int half_sel = lane >> 4;
    const int l16      = lane & 15;
    const int mrow     = mt * 16 + l16;     // A row (output channel)
    const int ncol     = nt * 16 + l16;     // B col (pixel in tile)
    const int kbaseA   = half_sel * 8;      // A: interleaved 8-K runs per lane-half
    const int kbaseB   = half_sel * 8;      // B: interleaved 8-K runs per lane-half
    const _Float16* arow  = wh + (size_t)mrow * CK;
    const _Float16* xbase = xt + (size_t)b * HW * CC;

    // gather roles: one wave -> 4 pixels, 8 lanes/pixel, 4 channels/lane
    const int fn  = tid >> 3;               // pixel 0..31
    const int fc4 = (tid & 7) * 4;          // channel quad within 32-chunk

    // one chunk fill: straight-line, 32-bit addressing, packed-f16 combine
    auto fill = [&](int it2, int sel2) {
        const int kcur = it2 >> 1;
        const int cc   = (it2 & 1) * 32 + fc4;    // channel base (32-bit)
        int4   ii = meta_i[fn * KT + kcur];
        float4 wv4 = meta_w[fn * KT + kcur];
        v4h p0 = *(const v4h*)(xbase + (ii.x + cc));
        v4h p1 = *(const v4h*)(xbase + (ii.y + cc));
        v4h p2 = *(const v4h*)(xbase + (ii.z + cc));
        v4h p3 = *(const v4h*)(xbase + (ii.w + cc));
        _Float16 h0 = (_Float16)wv4.x, h1 = (_Float16)wv4.y,
                 h2 = (_Float16)wv4.z, h3 = (_Float16)wv4.w;
        v4h h0v = {h0, h0, h0, h0}, h1v = {h1, h1, h1, h1};
        v4h h2v = {h2, h2, h2, h2}, h3v = {h3, h3, h3, h3};
        v4h rr = p0 * h0v + p1 * h1v + p2 * h2v + p3 * h3v;  // v_pk_fma_f16
        *(v4h*)&s_lds[sel2][fn * KCH + fc4] = rr;            // ds_store_b64
    };

    __syncthreads();                 // metadata visible
    fill(0, 0);                      // prologue: chunk 0 -> buffer 0
    __syncthreads();

    v8f acc = {};

    for (int it = 0; it < NIT; ++it) {
        const int sel = it & 1;
        const int ck0 = it * KCH;

        union { v16h v; v8h h[2]; } A, Bm;
        A.h[0]  = *(const v8h*)(arow + ck0 + kbaseA);        // K = kbase..kbase+7
        A.h[1]  = *(const v8h*)(arow + ck0 + kbaseA + 16);   // K = kbase+16..+23
        Bm.h[0] = *(const v8h*)(&s_lds[sel][ncol * KCH + kbaseB]);
        Bm.h[1] = *(const v8h*)(&s_lds[sel][ncol * KCH + kbaseB + 16]);

        if (it + 1 < NIT) fill(it + 1, sel ^ 1);             // overlap next gather

        acc = __builtin_amdgcn_wmma_f32_16x16x32_f16(
                  false, A.v, false, Bm.v, (short)0, acc, false, false);
        __syncthreads();             // single barrier per K-step
    }

    // ---- epilogue: D layout -> NCHW store (+bias) ----
    const int wo = wo0 + ncol;
    #pragma unroll
    for (int rr2 = 0; rr2 < 8; ++rr2) {
        int m = mt * 16 + half_sel * 8 + rr2;
        out[((size_t)(b * OO + m) * HH + ho) * WW + wo] = acc[rr2] + bias[m];
    }
}

// ---------------------------------------------------------------------------
extern "C" void kernel_launch(void* const* d_in, const int* in_sizes, int n_in,
                              void* d_out, int out_size, void* d_ws, size_t ws_size,
                              hipStream_t stream) {
    (void)in_sizes; (void)n_in; (void)out_size; (void)ws_size;
    const float* x      = (const float*)d_in[0];
    const float* offset = (const float*)d_in[1];
    const float* mask   = (const float*)d_in[2];
    const float* weight = (const float*)d_in[3];
    const float* bias   = (const float*)d_in[4];
    float* out = (float*)d_out;

    _Float16* wh = (_Float16*)d_ws;                         // 72 KB
    _Float16* xt = (_Float16*)((char*)d_ws + OO * CK * 2);  // 4.7 MB NHWC f16

    wprep<<<(OO * CK + 255) / 256, 256, 0, stream>>>(weight, wh);
    xprep<<<BB * (HW / 64), 256, 0, stream>>>(x, xt);
    dcn_wmma<<<BB * HH * (WW / NPIX), 256, 0, stream>>>(xt, offset, mask, wh, bias, out);
}
